// SPPMSE_81398220194175
// MI455X (gfx1250) — matrix operand order repaired and checked by reading, not compiled, not run on hardware
//
#include <hip/hip_runtime.h>
#include <hip/hip_bf16.h>
#include <math.h>

typedef __attribute__((ext_vector_type(16))) _Float16 v16h;
typedef __attribute__((ext_vector_type(8)))  _Float16 v8h;
typedef __attribute__((ext_vector_type(8)))  float    v8f;
typedef __attribute__((ext_vector_type(4)))  unsigned int v4u;
typedef __attribute__((ext_vector_type(8)))  int      v8i;
typedef __attribute__((ext_vector_type(4)))  int      v4i;

#define B_      64
#define NSAMP   128000
#define FRAME_  512
#define HOP_    256
#define PAD_    256
#define F_      257
#define T_      501
#define TTILES  32            // ceil(501/16)
#define NCT     16            // 16-wide f tiles per (cos|sin): f = 0..255
#define KSTEPS  16            // 512 / 32
#define FRAGS_PER_K 32        // 16 cos + 16 sin fragments per K-step
#define TAB_HALVES (KSTEPS*FRAGS_PER_K*512)   // 262144 f16 (pre-swizzled B frags)
#define TOTAL_ELEMS (B_*F_*T_)                // 8240448
#define KBLK_DWORDS (FRAGS_PER_K*512/2)       // 8192 dwords = 32 KB per K-step

// workspace layout (bytes)
#define OFF_TAB    0
#define OFF_NOISY  (TAB_HALVES*2)                 // 524288
#define OFF_NOISE  (OFF_NOISY + TOTAL_ELEMS*4)    // 33486080
#define OFF_PART   (OFF_NOISE + TOTAL_ELEMS*4)    // 66447872
#define NPART      1006                           // ceil(TOTAL_ELEMS/8192)

// ---------------------------------------------------------------------------
// Build the windowed DFT table directly in WMMA-B fragment order.
// B is K(=512, n) x N(=512: [cos f=0..255 | sin f=0..255]) in f16.
// Fragment (kt, jt): 32x16 tile. B layout used here (contiguous K, same as A):
//   lane l (0..15): element e -> (K = e,      N = l)
//   lane l+16:      element e -> (K = 16 + e, N = l)
// addr = (kt*32 + jt)*512 + lane*16 + e  -> one K-step is 32 KB contiguous.
// ---------------------------------------------------------------------------
__global__ __launch_bounds__(256) void build_tab(_Float16* __restrict__ tab) {
  const int idx = blockIdx.x * 256 + threadIdx.x;   // < TAB_HALVES
  const int frag = idx >> 9;       // kt*32 + jt
  const int eidx = idx & 511;
  const int lane = eidx >> 4;
  const int e    = eidx & 15;
  const int kt = frag >> 5;
  const int jt = frag & 31;
  const bool is_sin = (jt >= NCT);
  const int jl = jt & 15;
  const int f  = jl * 16 + (lane & 15);             // 0..255, always valid
  const int n  = kt * 32 + ((lane >> 4) * 16) + e;  // 0..511
  // sqrt-hann: w[n] = sin(pi*n/512); folded into DFT basis
  const float w = sinf((float)M_PI * (float)n * (1.0f / 512.0f));
  const int fn = (f * n) & 511;    // exact mod-512 argument reduction
  const float ang = 6.283185307179586f * (float)fn * (1.0f / 512.0f);
  const float val = w * (is_sin ? sinf(ang) : cosf(ang));
  tab[idx] = (_Float16)val;
}

// ---------------------------------------------------------------------------
// Issue one TDM 1-D bulk copy: 32 KB (one K-step of B fragments) global->LDS.
// D# per ISA ch.8: group0 {count=1, lds_addr, global_addr, type=2};
// group1 {data_size=4B, tensor_dim0=tile_dim0=8192 dwords, dim1=1, stride0}.
// Groups 2/3 zero (<=2D tensor).
// ---------------------------------------------------------------------------
__device__ __forceinline__ void tdm_load_kblock(const _Float16* gsrc, void* lds_dst) {
  const unsigned long long ga = (unsigned long long)(const void*)gsrc;
  const unsigned lds = (unsigned)(size_t)lds_dst;   // low 32 bits = LDS offset

  v4u g0 = {};
  g0[0] = 1u;                                          // count=1 (valid, user)
  g0[1] = lds;                                         // lds_addr [63:32]
  g0[2] = (unsigned)(ga & 0xffffffffu);                // global_addr [95:64]
  g0[3] = (unsigned)((ga >> 32) & 0x01ffffffu)         // global_addr [120:96]
        | 0x80000000u;                                 // type=2 ("image") [127:126]

  v8i g1 = {};
  g1[0] = 0x00020000;                                  // mask=0, data_size=2 (4B)
  g1[1] = (int)((KBLK_DWORDS & 0xffff) << 16);         // tensor_dim0 lo16
  g1[2] = (int)((KBLK_DWORDS >> 16) & 0xffff) | (1 << 16); // dim0 hi16 | tensor_dim1=1
  g1[3] = (int)((KBLK_DWORDS & 0xffff) << 16);         // tile_dim0 = 8192
  g1[4] = 1;                                           // tile_dim1 = 1
  g1[5] = KBLK_DWORDS;                                 // tensor_dim0_stride lo32
  g1[6] = 0;
  g1[7] = 0;

  v4i z4 = {};
#if __clang_major__ >= 23
  v8i z8 = {};
  __builtin_amdgcn_tensor_load_to_lds(g0, g1, z4, z4, z8, 0);
#else
  __builtin_amdgcn_tensor_load_to_lds(g0, g1, z4, z4, 0);
#endif
}

// ---------------------------------------------------------------------------
// STFT power via WMMA + TDM: out[b,f,t] = |sum_n x[t*HOP+n-PAD] * wdft[n,f]|^2
// for f = 0..255. One block = (batch b, 16-frame tile, signal select);
// 8 wave32 waves, each owning exactly two (cos,sin) tile pairs — no guards,
// so WMMA runs with full EXEC and accumulators never phi-copy.
// ---------------------------------------------------------------------------
__global__ __launch_bounds__(256) void stft_power_wmma(
    const float* __restrict__ sigA,      // input_sig
    const float* __restrict__ sigB,      // interference
    const _Float16* __restrict__ tab,
    float* __restrict__ outA,            // noisy power
    float* __restrict__ outB)            // noise power
{
  const int ttile = blockIdx.x;
  const int b     = blockIdx.y;
  const float* x  = (blockIdx.z == 0) ? sigA : sigB;
  float* outP     = (blockIdx.z == 0) ? outA : outB;

  __shared__ alignas(32) _Float16 As[16 * 512];           // 16 KB A tile
  __shared__ alignas(32) _Float16 Bs[FRAGS_PER_K * 512];  // 32 KB B frags (1 K-step)

  const int tid = threadIdx.x;
  for (int i = tid; i < 16 * 512; i += 256) {
    const int m = i >> 9;          // frame row within tile
    const int n = i & 511;         // sample within frame
    const int t = ttile * 16 + m;
    float v = 0.0f;
    if (t < T_) {
      int s = t * HOP_ + n - PAD_;
      if (s < 0) s = -s;                          // reflect (no edge repeat)
      if (s >= NSAMP) s = 2 * NSAMP - 2 - s;
      v = x[(size_t)b * NSAMP + s];
    }
    As[i] = (_Float16)v;
  }

  const int wave = tid >> 5;   // wave32: 0..7
  const int lane = tid & 31;
  const int hi   = lane >> 4;
  const int ln   = lane & 15;

  v8f accC0 = {};              // cos tile jt = wave
  v8f accS0 = {};              // sin tile jt = wave
  v8f accC1 = {};              // cos tile jt = wave + 8
  v8f accS1 = {};              // sin tile jt = wave + 8

  const unsigned offC0 = ((unsigned)(wave)            * 512 + lane * 16);
  const unsigned offS0 = ((unsigned)(NCT + wave)      * 512 + lane * 16);
  const unsigned offC1 = ((unsigned)(wave + 8)        * 512 + lane * 16);
  const unsigned offS1 = ((unsigned)(NCT + wave + 8)  * 512 + lane * 16);

  __syncthreads();

  for (int kt = 0; kt < KSTEPS; ++kt) {
    if (tid < 32) {            // wave 0 issues the TDM bulk copy for this K-step
      tdm_load_kblock(tab + (size_t)kt * (FRAGS_PER_K * 512), (void*)&Bs[0]);
      __builtin_amdgcn_s_wait_tensorcnt(0);
    }
    if (kt + 1 < KSTEPS) {
      __builtin_prefetch((const void*)(tab + (size_t)(kt + 1) * (FRAGS_PER_K * 512)), 0, 1);
    }
    __syncthreads();           // publish Bs to all waves

    // A fragment (16-bit A 16x32), contiguous K to match B table:
    //   lanes 0-15: e -> K=e; lanes 16-31: e -> K=16+e
    const _Float16* arow = &As[ln * 512 + kt * 32 + 16 * hi];
    union { v16h v; v8h h[2]; } ua;
    ua.h[0] = *(const v8h*)(arow);        // ds_load_b128
    ua.h[1] = *(const v8h*)(arow + 8);    // ds_load_b128
    const v16h a = ua.v;

    const v16h bc0 = *(const v16h*)&Bs[offC0];
    const v16h bs0 = *(const v16h*)&Bs[offS0];
    const v16h bc1 = *(const v16h*)&Bs[offC1];
    const v16h bs1 = *(const v16h*)&Bs[offS1];

    accC0 = __builtin_amdgcn_wmma_f32_16x16x32_f16(false, a, false, bc0, (short)0, accC0, false, false);
    accS0 = __builtin_amdgcn_wmma_f32_16x16x32_f16(false, a, false, bs0, (short)0, accS0, false, false);
    accC1 = __builtin_amdgcn_wmma_f32_16x16x32_f16(false, a, false, bc1, (short)0, accC1, false, false);
    accS1 = __builtin_amdgcn_wmma_f32_16x16x32_f16(false, a, false, bs1, (short)0, accS1, false, false);

    __syncthreads();           // all reads done before next TDM overwrites Bs
  }

  // D layout: lane l<16: VGPR r -> (M=r, N=l); lane l+16: (M=8+r, N=l)
  const int tbase = ttile * 16 + hi * 8;
  {
    const int f0 = wave * 16 + ln;                 // 0..127
    float* dst0 = outP + ((size_t)b * F_ + f0) * T_;
    const int f1 = (wave + 8) * 16 + ln;           // 128..255
    float* dst1 = outP + ((size_t)b * F_ + f1) * T_;
    #pragma unroll
    for (int r = 0; r < 8; ++r) {
      const int t = tbase + r;
      if (t < T_) {
        dst0[t] = accC0[r] * accC0[r] + accS0[r] * accS0[r];
        dst1[t] = accC1[r] * accC1[r] + accS1[r] * accS1[r];
      }
    }
  }
}

// ---------------------------------------------------------------------------
// Nyquist bin f=256: cos(pi*n)=(-1)^n, sin=0.
// power[b,256,t] = (sum_n (-1)^n * w[n] * x[t*HOP+n-PAD])^2
// ---------------------------------------------------------------------------
__global__ __launch_bounds__(256) void nyquist_power(
    const float* __restrict__ sigA, const float* __restrict__ sigB,
    float* __restrict__ outA, float* __restrict__ outB)
{
  const int idx = blockIdx.x * 256 + threadIdx.x;
  if (idx >= 2 * B_ * T_) return;
  const int z = idx / (B_ * T_);
  const int r = idx % (B_ * T_);
  const int b = r / T_;
  const int t = r % T_;
  const float* x = z ? sigB : sigA;
  float* o       = z ? outB : outA;
  float acc = 0.0f;
  for (int n = 0; n < FRAME_; ++n) {
    int s = t * HOP_ + n - PAD_;
    if (s < 0) s = -s;
    if (s >= NSAMP) s = 2 * NSAMP - 2 - s;
    const float w = sinf((float)M_PI * (float)n * (1.0f / 512.0f));
    const float v = x[(size_t)b * NSAMP + s] * w;
    acc += (n & 1) ? -v : v;
  }
  o[((size_t)b * F_ + 256) * T_ + t] = acc * acc;
}

// ---------------------------------------------------------------------------
// In-place exponential smoothing over t for each (b,f) row of noise power.
// ---------------------------------------------------------------------------
__global__ __launch_bounds__(256) void iir_scan(float* __restrict__ p) {
  const int idx = blockIdx.x * 256 + threadIdx.x;
  if (idx >= B_ * F_) return;
  float* row = p + (size_t)idx * T_;
  const float A   = 0.80073740f;       // exp(-HOP/(FS*0.072)) = exp(-256/1152)
  const float OmA = 1.0f - A;
  float v = row[0];                    // v0 = p[0] (row[0] already correct)
  for (int t = 1; t < T_; ++t) {
    v = A * v + OmA * row[t];
    row[t] = v;
  }
}

// ---------------------------------------------------------------------------
// MSE reduction, stage 1: 8192 elements per block -> one partial per block.
// ---------------------------------------------------------------------------
__global__ __launch_bounds__(256) void loss_partial(
    const float* __restrict__ est,
    const float* __restrict__ noisyP,
    const float* __restrict__ varN,
    float* __restrict__ part)
{
  __shared__ float red[256];
  const int tid = threadIdx.x;
  const size_t base = (size_t)blockIdx.x * 8192;
  float acc = 0.0f;
  for (int k = 0; k < 32; ++k) {
    const size_t e = base + (size_t)tid + (size_t)k * 256;
    if (e < (size_t)TOTAL_ELEMS) {
      const float np   = noisyP[e];
      const float vn   = varN[e];
      const float expo = -np / (vn + 1.1920929e-7f) * 0.96934657f; // XI/(1+XI)
      const float orac = 1.0f / (1.0f + 32.6227766f * expf(expo)); // (1+XI)
      const float d    = est[e] - orac;
      acc += d * d;
    }
  }
  red[tid] = acc;
  __syncthreads();
  for (int s = 128; s > 0; s >>= 1) {
    if (tid < s) red[tid] += red[tid + s];
    __syncthreads();
  }
  if (tid == 0) part[blockIdx.x] = red[0];
}

// Stage 2: fixed-order final reduce (deterministic, no float atomics).
__global__ __launch_bounds__(256) void loss_final(
    const float* __restrict__ part, float* __restrict__ out)
{
  __shared__ float red[256];
  const int tid = threadIdx.x;
  float acc = 0.0f;
  for (int j = tid; j < NPART; j += 256) acc += part[j];
  red[tid] = acc;
  __syncthreads();
  for (int s = 128; s > 0; s >>= 1) {
    if (tid < s) red[tid] += red[tid + s];
    __syncthreads();
  }
  if (tid == 0) out[0] = red[0] * (1.0f / (float)TOTAL_ELEMS);
}

// ---------------------------------------------------------------------------
extern "C" void kernel_launch(void* const* d_in, const int* in_sizes, int n_in,
                              void* d_out, int out_size, void* d_ws, size_t ws_size,
                              hipStream_t stream)
{
  const float* est  = (const float*)d_in[0];   // spp_estimate (B,1,F,T)
  const float* sig  = (const float*)d_in[1];   // input_sig    (B,1,N)
  const float* intf = (const float*)d_in[2];   // interference (B,1,N)

  char* ws = (char*)d_ws;
  _Float16* tab = (_Float16*)(ws + OFF_TAB);
  float* noisyP = (float*)(ws + OFF_NOISY);
  float* noiseP = (float*)(ws + OFF_NOISE);
  float* part   = (float*)(ws + OFF_PART);

  build_tab<<<TAB_HALVES / 256, 256, 0, stream>>>(tab);

  dim3 g(TTILES, B_, 2);
  stft_power_wmma<<<g, 256, 0, stream>>>(sig, intf, tab, noisyP, noiseP);

  nyquist_power<<<(2 * B_ * T_ + 255) / 256, 256, 0, stream>>>(sig, intf, noisyP, noiseP);

  iir_scan<<<(B_ * F_ + 255) / 256, 256, 0, stream>>>(noiseP);

  loss_partial<<<NPART, 256, 0, stream>>>(est, noisyP, noiseP, part);
  loss_final<<<1, 256, 0, stream>>>(part, (float*)d_out);
}
